// GCNConvLayer_6605659701694
// MI455X (gfx1250) — matrix operand order repaired and checked by reading, not compiled, not run on hardware
//
#include <hip/hip_runtime.h>
#include <hip/hip_bf16.h>
#include <stdint.h>

#define N_NODES 100000
#define N_EDGES 640000
#define D 128
#define N_TILES (N_NODES / 16)   // 6250, exact (no tail tiles)

typedef __attribute__((ext_vector_type(16))) __bf16 v16bf;
typedef __attribute__((ext_vector_type(2)))  __bf16 v2bf;
typedef __attribute__((ext_vector_type(8)))  float  v8f;

#if defined(__has_builtin)
#if __has_builtin(__builtin_amdgcn_cvt_pk_bf16_f32)
#define HAVE_CVT_PK_BF16 1
#endif
#endif

// ---------- helpers ----------
__device__ __forceinline__ uint32_t pack2bf(float lo, float hi) {
#ifdef HAVE_CVT_PK_BF16
    union { v2bf v; uint32_t u; } c;
    c.v = __builtin_amdgcn_cvt_pk_bf16_f32(lo, hi);
    return c.u;
#else
    uint32_t ul = __float_as_uint(lo);
    uint32_t uh = __float_as_uint(hi);
    uint32_t rl = (ul + 0x7FFFu + ((ul >> 16) & 1u)) >> 16;      // RNE
    uint32_t rh = (uh + 0x7FFFu + ((uh >> 16) & 1u)) >> 16;
    return rl | (rh << 16);
#endif
}

// Build 16x32 bf16 A-fragment for one lane from a row-major f32 row, with a
// per-row scale folded in (each lane holds exactly one row m = lane%16).
// Per ISA 7.12.2: lane L (m = L%16, half = L/16) holds
//   K = kb + group*16 + half*8 + w   (group=i/8, w=i%8)
__device__ __forceinline__ v16bf loadA(const float* __restrict__ row, int kb, int half, float s) {
    union { v16bf v; uint32_t u[8]; } f;
    const float* p0 = row + kb + half * 8;        // group 0: 8 contiguous floats
    const float* p1 = row + kb + 16 + half * 8;   // group 1
    float4 a0 = *(const float4*)(p0);
    float4 a1 = *(const float4*)(p0 + 4);
    float4 b0 = *(const float4*)(p1);
    float4 b1 = *(const float4*)(p1 + 4);
    f.u[0] = pack2bf(s * a0.x, s * a0.y); f.u[1] = pack2bf(s * a0.z, s * a0.w);
    f.u[2] = pack2bf(s * a1.x, s * a1.y); f.u[3] = pack2bf(s * a1.z, s * a1.w);
    f.u[4] = pack2bf(s * b0.x, s * b0.y); f.u[5] = pack2bf(s * b0.z, s * b0.w);
    f.u[6] = pack2bf(s * b1.x, s * b1.y); f.u[7] = pack2bf(s * b1.z, s * b1.w);
    return f.v;
}

// ---------- kernel 0: zero a float4 region ----------
__global__ void k_zero(float4* __restrict__ p, int n4) {
    int i = blockIdx.x * blockDim.x + threadIdx.x;
    int stride = gridDim.x * blockDim.x;
    float4 z = make_float4(0.f, 0.f, 0.f, 0.f);
    for (; i < n4; i += stride) p[i] = z;
}

// ---------- kernel 1: degrees (float atomics; exact for counts << 2^24) ----------
__global__ void k_deg(const int* __restrict__ src, const int* __restrict__ dst,
                      float* __restrict__ outdeg, float* __restrict__ indeg) {
    int e = blockIdx.x * blockDim.x + threadIdx.x;
    if (e >= N_EDGES) return;
    atomicAdd(&outdeg[src[e]], 1.0f);
    atomicAdd(&indeg[dst[e]], 1.0f);
}

// ---------- kernel 2: per-node normalization factors ----------
__global__ void k_scales(const float* __restrict__ indeg, const float* __restrict__ outdeg,
                         float* __restrict__ nsrc, float* __restrict__ alpha,
                         float* __restrict__ beta, float* __restrict__ gamma_,
                         float* __restrict__ delta) {
    int i = blockIdx.x * blockDim.x + threadIdx.x;
    if (i >= N_NODES) return;
    float id = indeg[i], od = outdeg[i];
    nsrc[i]  = rsqrtf(fmaxf(od, 1.0f));       // out_deg^-1/2 (clipped)
    alpha[i] = rsqrtf(fmaxf(id, 1.0f));       // norm_dst
    beta[i]  = 1.0f / (id + 1.0f);            // self-loop scale
    float g  = 1.0f / fmaxf(id, 1.0f);
    gamma_[i] = g;                            // edge-mean scale
    delta[i]  = id * g;                       // in_deg / max(in_deg,1): be weight
}

// ---------- kernel 3: fused edge scatter (32 lanes per edge, float4 per lane) ----------
__global__ void k_scatter(const float* __restrict__ nfeat, const float* __restrict__ efeat,
                          const int* __restrict__ src, const int* __restrict__ dst,
                          const float* __restrict__ nsrc,
                          float* __restrict__ S1, float* __restrict__ S2) {
    int tid = blockIdx.x * blockDim.x + threadIdx.x;
    int e = tid >> 5;
    if (e >= N_EDGES) return;
    int lane = tid & 31;
    int s = src[e], d = dst[e];
    float ns = nsrc[s];
    float4 nf = ((const float4*)(nfeat + (size_t)s * D))[lane];
    float4 ef = ((const float4*)(efeat + (size_t)e * D))[lane];
    float* p1 = S1 + (size_t)d * D + lane * 4;
    float* p2 = S2 + (size_t)d * D + lane * 4;
    atomicAdd(p1 + 0, ns * nf.x); atomicAdd(p1 + 1, ns * nf.y);
    atomicAdd(p1 + 2, ns * nf.z); atomicAdd(p1 + 3, ns * nf.w);
    atomicAdd(p2 + 0, ef.x); atomicAdd(p2 + 1, ef.y);
    atomicAdd(p2 + 2, ef.z); atomicAdd(p2 + 3, ef.w);
}

// ---------- kernel 4: 3-way WMMA GEMM + combine ----------
// S1 aliases `out`: each wave reads its 16 rows of S1 before storing them.
// Row scales are folded into the (f32) A operand pre-conversion; biases are
// folded into the accumulator init, so all 3 branches chain through WMMA C.
__global__ __launch_bounds__(128) void k_gemm(
    const float* __restrict__ S1, const float* __restrict__ nfeat, const float* __restrict__ S2,
    const float* __restrict__ W, const float* __restrict__ We,
    const float* __restrict__ b, const float* __restrict__ be,
    const float* __restrict__ alpha, const float* __restrict__ beta,
    const float* __restrict__ gamma_, const float* __restrict__ delta,
    float* __restrict__ out)
{
    // B fragments pre-swizzled: [mat][kc][nt][lane][8 uints] = 16384 uints = 64 KB LDS.
    // Lane layout (32x16 bf16 B): n = lane%16, K = kc*32 + (i/8)*16 + (lane/16)*8 + i%8.
    __shared__ uint32_t Bf[16384];

    int t = threadIdx.x;
    for (int q = t; q < 16384; q += 128) {
        int j   = q & 7;            // uint within lane (elems 2j, 2j+1)
        int ln  = (q >> 3) & 31;
        int nt  = (q >> 8) & 7;
        int kc  = (q >> 11) & 3;
        int mat = (q >> 13) & 1;
        int n   = nt * 16 + (ln & 15);
        int K0  = kc * 32 + ((j >> 2) << 4) + ((ln >> 4) << 3) + 2 * (j & 3);
        const float* Wm = mat ? We : W;           // W is [K][N] row-major
        Bf[q] = pack2bf(Wm[K0 * D + n], Wm[(K0 + 1) * D + n]);
    }
    __syncthreads();

    int wave = t >> 5;
    int lane = t & 31;
    int tile = blockIdx.x * 4 + wave;
    if (tile >= N_TILES) return;
    int mbase = tile * 16;
    int lm = lane & 15;
    int half = lane >> 4;
    int m = mbase + lm;                 // A-fragment row for this lane
    int rbase = mbase + half * 8;       // C-fragment rows: rbase..rbase+7

    // per-branch A row scales (one row per lane)
    float sc_a[3];
    sc_a[0] = alpha[m];                 // (S1@W)  * norm_dst
    sc_a[1] = beta[m];                  // (x@W)   / (in_deg+1)
    sc_a[2] = gamma_[m];                // (S2@We) / max(in_deg,1)

    // bias terms follow the C layout (rows rbase..rbase+7)
    float bt[8], dl[8];
    #pragma unroll
    for (int r = 0; r < 8; ++r) {
        bt[r] = beta[rbase + r];
        dl[r] = delta[rbase + r];
    }

    // accumulators initialized with bias: b[n]*(1+beta[m]) + be[n]*delta[m]
    v8f acc[8];
    #pragma unroll
    for (int nt = 0; nt < 8; ++nt) {
        int n = nt * 16 + lm;
        float bn = b[n], ben = be[n];
        #pragma unroll
        for (int r = 0; r < 8; ++r)
            acc[nt][r] = bn * (1.f + bt[r]) + ben * dl[r];
    }

    const float* srcs[3] = { S1 + (size_t)m * D, nfeat + (size_t)m * D, S2 + (size_t)m * D };

    #pragma unroll
    for (int br = 0; br < 3; ++br) {
        const float* rp = srcs[br];
        float s = sc_a[br];
        v16bf a0 = loadA(rp, 0, half, s);
        v16bf a1 = loadA(rp, 32, half, s);
        v16bf a2 = loadA(rp, 64, half, s);
        v16bf a3 = loadA(rp, 96, half, s);
        int mat = (br == 2) ? 1 : 0;
        #pragma unroll
        for (int nt = 0; nt < 8; ++nt) {
            #pragma unroll
            for (int kc = 0; kc < 4; ++kc) {
                union { v16bf v; uint4 q[2]; } bfrag;
                const uint4* p = (const uint4*)&Bf[(((mat * 4 + kc) * 8 + nt) * 32 + lane) * 8];
                bfrag.q[0] = p[0];
                bfrag.q[1] = p[1];
                v16bf av = (kc == 0) ? a0 : (kc == 1) ? a1 : (kc == 2) ? a2 : a3;
                acc[nt] = __builtin_amdgcn_wmma_f32_16x16x32_bf16(
                            false, av, false, bfrag.v, (short)0, acc[nt], false, false);
            }
        }
    }

    // store (reads of S1 rows for this tile happened above; rows disjoint per wave)
    #pragma unroll
    for (int nt = 0; nt < 8; ++nt) {
        int n = nt * 16 + lm;
        #pragma unroll
        for (int r = 0; r < 8; ++r)
            out[(size_t)(rbase + r) * D + n] = acc[nt][r];
    }
}

// ---------- launcher ----------
extern "C" void kernel_launch(void* const* d_in, const int* in_sizes, int n_in,
                              void* d_out, int out_size, void* d_ws, size_t ws_size,
                              hipStream_t stream) {
    const float* nfeat = (const float*)d_in[0];
    const float* efeat = (const float*)d_in[1];
    const int*   src   = (const int*)d_in[2];
    const int*   dst   = (const int*)d_in[3];
    const float* W     = (const float*)d_in[4];
    const float* b     = (const float*)d_in[5];
    const float* We    = (const float*)d_in[6];
    const float* be    = (const float*)d_in[7];
    float* out = (float*)d_out;

    char* ws = (char*)d_ws;
    size_t off = 0;
    float* S2 = (float*)(ws + off);      off += (size_t)N_NODES * D * sizeof(float);
    float* indeg  = (float*)(ws + off);  off += (size_t)N_NODES * sizeof(float);
    float* outdeg = (float*)(ws + off);  off += (size_t)N_NODES * sizeof(float);
    float* nsrc   = (float*)(ws + off);  off += (size_t)N_NODES * sizeof(float);
    float* alpha  = (float*)(ws + off);  off += (size_t)N_NODES * sizeof(float);
    float* beta   = (float*)(ws + off);  off += (size_t)N_NODES * sizeof(float);
    float* gamma_ = (float*)(ws + off);  off += (size_t)N_NODES * sizeof(float);
    float* delta  = (float*)(ws + off);  off += (size_t)N_NODES * sizeof(float);

    const int ZB = 1024;
    int nd4 = N_NODES * D / 4;
    k_zero<<<(nd4 + ZB * 4 - 1) / (ZB * 4), ZB, 0, stream>>>((float4*)out, nd4);   // S1 accumulator
    k_zero<<<(nd4 + ZB * 4 - 1) / (ZB * 4), ZB, 0, stream>>>((float4*)S2, nd4);
    int ndeg4 = (2 * N_NODES) / 4;                                                 // indeg+outdeg contiguous
    k_zero<<<(ndeg4 + ZB - 1) / ZB, ZB, 0, stream>>>((float4*)indeg, ndeg4);

    k_deg<<<(N_EDGES + 255) / 256, 256, 0, stream>>>(src, dst, outdeg, indeg);
    k_scales<<<(N_NODES + 255) / 256, 256, 0, stream>>>(indeg, outdeg, nsrc, alpha, beta, gamma_, delta);

    int sthreads = N_EDGES * 32;
    k_scatter<<<(sthreads + 255) / 256, 256, 0, stream>>>(nfeat, efeat, src, dst, nsrc, out, S2);

    k_gemm<<<(N_TILES + 3) / 4, 128, 0, stream>>>(out, nfeat, S2, W, We, b, be,
                                                  alpha, beta, gamma_, delta, out);
}
